// BigBirdBoxHead_91216515432736
// MI455X (gfx1250) — matrix operand order repaired and checked by reading, not compiled, not run on hardware
//
#include <hip/hip_runtime.h>
#include <hip/hip_bf16.h>
#include <math.h>

typedef __bf16 bf16_t;
typedef __attribute__((ext_vector_type(16))) __bf16 v16bf;
typedef __attribute__((ext_vector_type(8)))  __bf16 v8bf;
typedef __attribute__((ext_vector_type(8)))  float  v8f;
typedef __attribute__((ext_vector_type(4)))  unsigned int v4u;

#define BDIM 256
#define SEQ  49
#define DMODEL 1024
#define NHEAD 2
#define HDIM 512
#define MTOK (BDIM * SEQ)     // 12544 token rows

// async global->LDS copy (ASYNCcnt); CDNA5 ISA 15.18.3 op 98, GV mode
__device__ __forceinline__ void async_copy_b128(unsigned ldsOff, const bf16_t* g) {
  asm volatile("global_load_async_to_lds_b128 %0, %1, off"
               :: "v"(ldsOff), "v"(g) : "memory");
}
__device__ __forceinline__ void wait_async0() {
  asm volatile("s_wait_asynccnt 0" ::: "memory");
}

// ---------------- generic batched bf16 GEMM with WMMA ----------------
// C[M,N] = A[M,K] (bf16, row-major, K contiguous) x B'[N,K] (bf16, K contiguous)
// 128x128 block, 8 wave32s (4 x 2), wave tile 32x64 -> 8 WMMA / K-step.
// Double-buffered LDS. Aligned path: GLOBAL_LOAD_ASYNC_TO_LDS_B128 pipeline.
// Ragged path: register-staged loads with zero fill.
// epi: 0 bf16 plain | 1 bf16 bias+gelu | 2 f32 resid | 3 f32 resid+bias
//      4 f32 bias+relu | 5 f32 plain
#define GBM 128
#define GBN 128
#define GKB 32
#define LDSROW 40                        // 32 + 8 pad (80B rows, 16B aligned)
#define TILESZ (128 * LDSROW)

template <bool MAL, bool NAL>
__global__ __launch_bounds__(256) void gemm_bf16_wmma(
    const bf16_t* __restrict__ A, long lda, long aBo, long aBi,
    const bf16_t* __restrict__ Bm, long ldb, long bBo, long bBi,
    void* __restrict__ Cp, long ldc, long cBo, long cBi,
    const float* __restrict__ bias,
    const float* __restrict__ resid,
    int M, int N, int K, int batchInner, int epi)
{
  __shared__ alignas(16) bf16_t As[2 * TILESZ];
  __shared__ alignas(16) bf16_t Bs[2 * TILESZ];

  const int tid  = threadIdx.x;
  const int lane = tid & 31;
  const int wave = tid >> 5;
  const int wm   = wave >> 1;   // 0..3  (M direction, 32 rows each)
  const int wn   = wave & 1;    // 0..1  (N direction, 64 cols each)
  const int half = lane >> 4;
  const int l15  = lane & 15;

  const int rowBlock = blockIdx.x * GBM;
  const int colBlock = blockIdx.y * GBN;

  const int z  = blockIdx.z;
  const int zo = z / batchInner;
  const int zi = z % batchInner;
  A  += (long)zo * aBo + (long)zi * aBi;
  Bm += (long)zo * bBo + (long)zi * bBi;
  const long cOff = (long)zo * cBo + (long)zi * cBi;

  v8f acc[2][4];
#pragma unroll
  for (int tm = 0; tm < 2; ++tm)
#pragma unroll
    for (int tn = 0; tn < 4; ++tn)
#pragma unroll
      for (int e = 0; e < 8; ++e) acc[tm][tn][e] = 0.0f;

  const int KT = K / GKB;

  // per-K-step compute from LDS buffer `buf`
  auto computeStep = [&](int buf) {
    const bf16_t* Ab = As + buf * TILESZ;
    const bf16_t* Bb = Bs + buf * TILESZ;
    v16bf aF[2], bF[4];
#pragma unroll
    for (int tm = 0; tm < 2; ++tm) {
      int r  = wm * 32 + tm * 16 + l15;
      int k0 = half * 8;
      v8bf lo = *(const v8bf*)(Ab + r * LDSROW + k0);
      v8bf hi = *(const v8bf*)(Ab + r * LDSROW + k0 + 16);
      aF[tm] = __builtin_shufflevector(lo, hi, 0,1,2,3,4,5,6,7,8,9,10,11,12,13,14,15);
    }
#pragma unroll
    for (int tn = 0; tn < 4; ++tn) {
      int n  = wn * 64 + tn * 16 + l15;
      int k0 = half * 8;
      v8bf lo = *(const v8bf*)(Bb + n * LDSROW + k0);
      v8bf hi = *(const v8bf*)(Bb + n * LDSROW + k0 + 16);
      bF[tn] = __builtin_shufflevector(lo, hi, 0,1,2,3,4,5,6,7,8,9,10,11,12,13,14,15);
    }
#pragma unroll
    for (int tm = 0; tm < 2; ++tm)
#pragma unroll
      for (int tn = 0; tn < 4; ++tn)
        acc[tm][tn] = __builtin_amdgcn_wmma_f32_16x16x32_bf16(
            false, aF[tm], false, bF[tn], (short)0, acc[tm][tn], false, false);
  };

  if constexpr (MAL && NAL) {
    // ---------- async global->LDS pipeline (no VGPR staging) ----------
    const unsigned asBase = (unsigned)(size_t)As;
    const unsigned bsBase = (unsigned)(size_t)Bs;
    auto issueAsync = [&](int buf, int kk) {
#pragma unroll
      for (int j = 0; j < 2; ++j) {
        int slot = tid + j * 256;          // 512 x 16B slots per 128x32 tile
        int r  = slot >> 2;
        int kc = (slot & 3) * 8;
        unsigned ldsOff = (unsigned)((buf * TILESZ + r * LDSROW + kc) * 2);
        async_copy_b128(asBase + ldsOff, A  + (long)(rowBlock + r) * lda + kk + kc);
        async_copy_b128(bsBase + ldsOff, Bm + (long)(colBlock + r) * ldb + kk + kc);
      }
      if (kk + 2 * GKB < K) {              // speculative prefetch of K+2 chunk
        int r0 = tid >> 2, kc0 = (tid & 3) * 8;
        __builtin_prefetch((const void*)(A  + (long)(rowBlock + r0) * lda + kk + 2 * GKB + kc0), 0, 1);
        __builtin_prefetch((const void*)(Bm + (long)(colBlock + r0) * ldb + kk + 2 * GKB + kc0), 0, 1);
      }
    };
    issueAsync(0, 0);
    for (int kt = 0; kt < KT; ++kt) {
      wait_async0();                       // my buffer-kt copies landed in LDS
      __syncthreads();                     // everyone's landed; buffer kt+1 free (WAR)
      if (kt + 1 < KT) issueAsync((kt + 1) & 1, (kt + 1) * GKB);
      computeStep(kt & 1);
    }
  } else {
    // ---------- register-staged path with zero-fill for ragged edges ----------
    v4u aReg[2], bReg[2];
    auto loadG = [&](int kk) {
#pragma unroll
      for (int j = 0; j < 2; ++j) {
        int slot = tid + j * 256;
        int r  = slot >> 2;
        int kc = (slot & 3) * 8;
        v4u zv = {0u, 0u, 0u, 0u};
        int gr = rowBlock + r;
        int gn = colBlock + r;
        aReg[j] = (MAL || gr < M) ? *(const v4u*)(A  + (long)gr * lda + kk + kc) : zv;
        bReg[j] = (NAL || gn < N) ? *(const v4u*)(Bm + (long)gn * ldb + kk + kc) : zv;
      }
    };
    auto storeL = [&](int buf) {
#pragma unroll
      for (int j = 0; j < 2; ++j) {
        int slot = tid + j * 256;
        int r  = slot >> 2;
        int kc = (slot & 3) * 8;
        *(v4u*)(As + buf * TILESZ + r * LDSROW + kc) = aReg[j];
        *(v4u*)(Bs + buf * TILESZ + r * LDSROW + kc) = bReg[j];
      }
    };
    loadG(0);
    storeL(0);
    for (int kt = 0; kt < KT; ++kt) {
      __syncthreads();
      if (kt + 1 < KT) loadG((kt + 1) * GKB);
      computeStep(kt & 1);
      if (kt + 1 < KT) storeL((kt + 1) & 1);
    }
  }

  // ---- epilogue: D layout: VGPR r -> row (r + half*8), col = l15 ----
#pragma unroll
  for (int tm = 0; tm < 2; ++tm) {
#pragma unroll
    for (int tn = 0; tn < 4; ++tn) {
      int col = colBlock + wn * 64 + tn * 16 + l15;
      if (!NAL && col >= N) continue;
      float bv = bias ? bias[col] : 0.0f;
#pragma unroll
      for (int r = 0; r < 8; ++r) {
        int row = rowBlock + wm * 32 + tm * 16 + half * 8 + r;
        if (!MAL && row >= M) continue;
        float vA = acc[tm][tn][r];
        long idx = cOff + (long)row * ldc + col;
        switch (epi) {
          case 0: ((bf16_t*)Cp)[idx] = (bf16_t)vA; break;
          case 1: {
            float g = vA + bv;
            float t = 0.5f * g * (1.0f + tanhf(0.7978845608028654f * (g + 0.044715f * g * g * g)));
            ((bf16_t*)Cp)[idx] = (bf16_t)t;
          } break;
          case 2: ((float*)Cp)[idx] = resid[idx] + vA; break;
          case 3: ((float*)Cp)[idx] = resid[idx] + vA + bv; break;
          case 4: ((float*)Cp)[idx] = fmaxf(vA + bv, 0.0f); break;
          default: ((float*)Cp)[idx] = vA; break;
        }
      }
    }
  }
}

// ---------------- weight prep: fp32 [R,C] -> bf16 [C,R] ----------------
__global__ __launch_bounds__(256) void wprep(const float* __restrict__ src,
                                             bf16_t* __restrict__ dst,
                                             int R, int Ccols)
{
  __shared__ float t[32][33];
  int tx = threadIdx.x & 31;
  int ty = threadIdx.x >> 5;
  int k0 = blockIdx.x * 32;
  int n0 = blockIdx.y * 32;
#pragma unroll
  for (int j = 0; j < 4; ++j)
    t[ty + j * 8][tx] = src[(long)(k0 + ty + j * 8) * Ccols + n0 + tx];
  __syncthreads();
#pragma unroll
  for (int j = 0; j < 4; ++j)
    dst[(long)(n0 + ty + j * 8) * R + k0 + tx] = (bf16_t)t[tx][ty + j * 8];
}

// ---------------- embed: x[b,d,s] + pos[s,d] -> h[b*S+s, d] fp32 ----------------
__global__ __launch_bounds__(256) void embed_k(const float* __restrict__ x,
                                               const float* __restrict__ pos,
                                               float* __restrict__ h)
{
  long idx = (long)blockIdx.x * 256 + threadIdx.x;   // MTOK*D
  int d = (int)(idx & 1023);
  long row = idx >> 10;
  int s = (int)(row % SEQ);
  long b = row / SEQ;
  h[idx] = x[b * (long)(DMODEL * SEQ) + (long)d * SEQ + s] + pos[(long)s * DMODEL + d];
}

// ---------------- LayerNorm over rows of D=1024, write bf16 ----------------
__global__ __launch_bounds__(256) void ln_rows(const float* __restrict__ x,
                                               const float* __restrict__ sc,
                                               const float* __restrict__ bi,
                                               bf16_t* __restrict__ out)
{
  const int D = DMODEL;
  long row = blockIdx.x;
  const float* xr = x + row * D;
  int tid = threadIdx.x;
  float v[4]; float sum = 0.f, sq = 0.f;
#pragma unroll
  for (int j = 0; j < 4; ++j) { v[j] = xr[tid + j * 256]; sum += v[j]; sq += v[j] * v[j]; }
#pragma unroll
  for (int off = 16; off; off >>= 1) {
    sum += __shfl_xor(sum, off, 32);
    sq  += __shfl_xor(sq,  off, 32);
  }
  __shared__ float red[2][8];
  int wv = tid >> 5, lane = tid & 31;
  if (lane == 0) { red[0][wv] = sum; red[1][wv] = sq; }
  __syncthreads();
  float ts = 0.f, tq = 0.f;
#pragma unroll
  for (int i = 0; i < 8; ++i) { ts += red[0][i]; tq += red[1][i]; }
  float mean = ts * (1.0f / 1024.0f);
  float var  = tq * (1.0f / 1024.0f) - mean * mean;
  float inv  = rsqrtf(var + 1e-5f);
  bf16_t* o = out + row * D;
#pragma unroll
  for (int j = 0; j < 4; ++j) {
    int c = tid + j * 256;
    o[c] = (bf16_t)((v[j] - mean) * inv * sc[c] + bi[c]);
  }
}

// ------- V transpose: qkv[b*S+k, 2048 + h*HD+hd] -> vt[b][h][hd][k64] (K pad 64) -------
__global__ __launch_bounds__(256) void vtrans(const bf16_t* __restrict__ qkv,
                                              bf16_t* __restrict__ vt)
{
  long i = (long)blockIdx.x * 256 + threadIdx.x;   // B*H*HD*64
  int kk = (int)(i & 63);
  long r = i >> 6;
  int hd = (int)(r & (HDIM - 1));
  r >>= 9;
  int hh = (int)(r & 1);
  long bb = r >> 1;
  bf16_t val = (bf16_t)0.0f;
  if (kk < SEQ) val = qkv[(bb * SEQ + kk) * (3 * DMODEL) + 2 * DMODEL + hh * HDIM + hd];
  vt[i] = val;
}

// ------- softmax over 49 logits, scaled, write bf16 padded to 64 -------
__global__ __launch_bounds__(256) void softmax_k(const float* __restrict__ sc,
                                                 bf16_t* __restrict__ attn)
{
  int wv = threadIdx.x >> 5, lane = threadIdx.x & 31;
  long row = (long)blockIdx.x * 8 + wv;            // B*H*SEQ rows
  const float* p = sc + row * 64;
  const float scale = 0.044194173824159216f;       // 1/sqrt(512)
  float x0 = (lane      < SEQ) ? p[lane]      * scale : -3.0e38f;
  float x1 = (lane + 32 < SEQ) ? p[lane + 32] * scale : -3.0e38f;
  float m = fmaxf(x0, x1);
#pragma unroll
  for (int off = 16; off; off >>= 1) m = fmaxf(m, __shfl_xor(m, off, 32));
  float e0 = (lane      < SEQ) ? __expf(x0 - m) : 0.0f;
  float e1 = (lane + 32 < SEQ) ? __expf(x1 - m) : 0.0f;
  float s = e0 + e1;
#pragma unroll
  for (int off = 16; off; off >>= 1) s += __shfl_xor(s, off, 32);
  float inv = 1.0f / s;
  bf16_t* o = attn + row * 64;
  o[lane]      = (bf16_t)(e0 * inv);
  o[lane + 32] = (bf16_t)(e1 * inv);
}

// ------- split-K reduction for the head: out = relu(sum_z parts[z] + bias) -------
#define KSPLIT 16
__global__ __launch_bounds__(256) void head_reduce(const float* __restrict__ parts,
                                                   const float* __restrict__ bias,
                                                   float* __restrict__ out)
{
  long idx = (long)blockIdx.x * 256 + threadIdx.x;  // 256*1024
  int n = (int)(idx & 1023);
  float s = bias[n];
#pragma unroll
  for (int zp = 0; zp < KSPLIT; ++zp) s += parts[(long)zp * (BDIM * 1024) + idx];
  out[idx] = fmaxf(s, 0.0f);
}

// =====================================================================
static inline void launch_gemm(hipStream_t stream,
    const bf16_t* A, long lda, long aBo, long aBi,
    const bf16_t* B, long ldb, long bBo, long bBi,
    void* C, long ldc, long cBo, long cBi,
    const float* bias, const float* resid,
    int M, int N, int K, int nz, int batchInner, int epi)
{
  dim3 grid((M + GBM - 1) / GBM, (N + GBN - 1) / GBN, nz);
  bool mal = (M % GBM) == 0, nal = (N % GBN) == 0;
  if (mal && nal)
    gemm_bf16_wmma<true, true><<<grid, dim3(256), 0, stream>>>(A, lda, aBo, aBi, B, ldb, bBo, bBi, C, ldc, cBo, cBi, bias, resid, M, N, K, batchInner, epi);
  else if (mal)
    gemm_bf16_wmma<true, false><<<grid, dim3(256), 0, stream>>>(A, lda, aBo, aBi, B, ldb, bBo, bBi, C, ldc, cBo, cBi, bias, resid, M, N, K, batchInner, epi);
  else if (nal)
    gemm_bf16_wmma<false, true><<<grid, dim3(256), 0, stream>>>(A, lda, aBo, aBi, B, ldb, bBo, bBi, C, ldc, cBo, cBi, bias, resid, M, N, K, batchInner, epi);
  else
    gemm_bf16_wmma<false, false><<<grid, dim3(256), 0, stream>>>(A, lda, aBo, aBi, B, ldb, bBo, bBi, C, ldc, cBo, cBi, bias, resid, M, N, K, batchInner, epi);
}

extern "C" void kernel_launch(void* const* d_in, const int* in_sizes, int n_in,
                              void* d_out, int out_size, void* d_ws, size_t ws_size,
                              hipStream_t stream) {
  const float* x      = (const float*)d_in[0];
  const float* pos    = (const float*)d_in[1];
  const float* ln1_s  = (const float*)d_in[2];
  const float* ln1_b  = (const float*)d_in[3];
  const float* wq     = (const float*)d_in[4];
  const float* wk     = (const float*)d_in[5];
  const float* wv     = (const float*)d_in[6];
  const float* wo     = (const float*)d_in[7];
  const float* ln2_s  = (const float*)d_in[8];
  const float* ln2_b  = (const float*)d_in[9];
  const float* w1     = (const float*)d_in[10];
  const float* b1     = (const float*)d_in[11];
  const float* w2     = (const float*)d_in[12];
  const float* b2     = (const float*)d_in[13];
  const float* lnf_s  = (const float*)d_in[14];
  const float* lnf_b  = (const float*)d_in[15];
  const float* head_w = (const float*)d_in[16];
  const float* head_b = (const float*)d_in[17];
  float* out = (float*)d_out;

  // ---- workspace layout ----
  char* base = (char*)d_ws;
  size_t off = 0;
  auto alloc = [&](size_t elems, size_t esz) -> char* {
    char* p = base + off;
    off += (elems * esz + 255) & ~(size_t)255;
    return p;
  };
  const size_t WSQ = 1024 * 1024;
  bf16_t* qkvT[2]; bf16_t* woT[2]; bf16_t* w1T[2]; bf16_t* w2T[2];
  for (int i = 0; i < 2; ++i) {
    qkvT[i] = (bf16_t*)alloc(3 * WSQ, 2);      // packed [wqT | wkT | wvT]
    woT[i]  = (bf16_t*)alloc(WSQ, 2);
    w1T[i]  = (bf16_t*)alloc(WSQ, 2);
    w2T[i]  = (bf16_t*)alloc(WSQ, 2);
  }
  bf16_t* headT = (bf16_t*)alloc((size_t)1024 * 50176, 2);
  float*  h     = (float*) alloc((size_t)MTOK * DMODEL, 4);
  bf16_t* hn    = (bf16_t*)alloc((size_t)MTOK * DMODEL, 2);
  bf16_t* qkv   = (bf16_t*)alloc((size_t)MTOK * 3 * DMODEL, 2);
  bf16_t* vt    = (bf16_t*)alloc((size_t)BDIM * NHEAD * HDIM * 64, 2);
  float*  scs   = (float*) alloc((size_t)BDIM * NHEAD * SEQ * 64, 4);
  bf16_t* attn  = (bf16_t*)alloc((size_t)BDIM * NHEAD * SEQ * 64, 2);
  bf16_t* ctx   = (bf16_t*)alloc((size_t)MTOK * DMODEL, 2);
  bf16_t* ffh   = (bf16_t*)alloc((size_t)MTOK * DMODEL, 2);
  float*  parts = (float*) alloc((size_t)KSPLIT * BDIM * 1024, 4);
  (void)ws_size; (void)in_sizes; (void)n_in; (void)out_size;

  // ---- weight prep (fp32 -> bf16, [K,N] -> [N,K]) ----
  for (int i = 0; i < 2; ++i) {
    wprep<<<dim3(32, 32), 256, 0, stream>>>(wq + i * WSQ, qkvT[i],            1024, 1024);
    wprep<<<dim3(32, 32), 256, 0, stream>>>(wk + i * WSQ, qkvT[i] + WSQ,      1024, 1024);
    wprep<<<dim3(32, 32), 256, 0, stream>>>(wv + i * WSQ, qkvT[i] + 2 * WSQ,  1024, 1024);
    wprep<<<dim3(32, 32), 256, 0, stream>>>(wo + i * WSQ, woT[i], 1024, 1024);
    wprep<<<dim3(32, 32), 256, 0, stream>>>(w1 + i * WSQ, w1T[i], 1024, 1024);
    wprep<<<dim3(32, 32), 256, 0, stream>>>(w2 + i * WSQ, w2T[i], 1024, 1024);
  }
  wprep<<<dim3(1568, 32), 256, 0, stream>>>(head_w, headT, 50176, 1024);

  // ---- embed ----
  embed_k<<<dim3((MTOK * DMODEL) / 256), 256, 0, stream>>>(x, pos, h);

  const long QKV_LD = 3 * DMODEL;
  const long TOKQ_B = (long)SEQ * QKV_LD;      // per-batch row stride in qkv
  const long TOK_B  = (long)SEQ * DMODEL;      // per-batch row stride in token stream
  for (int i = 0; i < 2; ++i) {
    // --- attention block ---
    ln_rows<<<dim3(MTOK), 256, 0, stream>>>(h, ln1_s + i * DMODEL, ln1_b + i * DMODEL, hn);
    // fused QKV projection: [12544,1024] x [3072,1024]' -> [12544,3072]
    launch_gemm(stream, hn, DMODEL, 0, 0, qkvT[i], DMODEL, 0, 0, qkv, QKV_LD, 0, 0,
                nullptr, nullptr, MTOK, 3 * DMODEL, DMODEL, 1, 1, 0);
    vtrans<<<dim3((BDIM * NHEAD * HDIM * 64) / 256), 256, 0, stream>>>(qkv, vt);
    // scores[b,h,q,k], batched z = b*2+h; Q at col 0, K at col 1024 of qkv
    launch_gemm(stream, qkv, QKV_LD, TOKQ_B, HDIM, qkv + DMODEL, QKV_LD, TOKQ_B, HDIM,
                scs, 64, (long)NHEAD * SEQ * 64, (long)SEQ * 64,
                nullptr, nullptr, SEQ, SEQ, HDIM, BDIM * NHEAD, NHEAD, 5);
    softmax_k<<<dim3((BDIM * NHEAD * SEQ) / 8), 256, 0, stream>>>(scs, attn);
    // ctx[q, h*512+hd] = attn[q,k] @ vt[hd,k]  (K padded to 64)
    launch_gemm(stream, attn, 64, (long)NHEAD * SEQ * 64, (long)SEQ * 64,
                vt, 64, (long)NHEAD * HDIM * 64, (long)HDIM * 64,
                ctx, DMODEL, TOK_B, HDIM,
                nullptr, nullptr, SEQ, HDIM, 64, BDIM * NHEAD, NHEAD, 0);
    // h += ctx @ Wo
    launch_gemm(stream, ctx, DMODEL, 0, 0, woT[i], DMODEL, 0, 0, h, DMODEL, 0, 0,
                nullptr, h, MTOK, DMODEL, DMODEL, 1, 1, 2);
    // --- FFN block ---
    ln_rows<<<dim3(MTOK), 256, 0, stream>>>(h, ln2_s + i * DMODEL, ln2_b + i * DMODEL, hn);
    launch_gemm(stream, hn, DMODEL, 0, 0, w1T[i], DMODEL, 0, 0, ffh, DMODEL, 0, 0,
                b1 + i * DMODEL, nullptr, MTOK, DMODEL, DMODEL, 1, 1, 1);
    launch_gemm(stream, ffh, DMODEL, 0, 0, w2T[i], DMODEL, 0, 0, h, DMODEL, 0, 0,
                b2 + i * DMODEL, h, MTOK, DMODEL, DMODEL, 1, 1, 3);
  }

  // ---- final LN + split-K head (hn viewed flat as [256, 50176]) ----
  ln_rows<<<dim3(MTOK), 256, 0, stream>>>(h, lnf_s, lnf_b, hn);
  const long KS = (SEQ * DMODEL) / KSPLIT;     // 3136, multiple of 32
  launch_gemm(stream, hn, (long)SEQ * DMODEL, KS, 0, headT, (long)SEQ * DMODEL, KS, 0,
              parts, 1024, (long)BDIM * 1024, 0,
              nullptr, nullptr, BDIM, 1024, (int)KS, KSPLIT, 1, 5);
  head_reduce<<<dim3((BDIM * 1024) / 256), 256, 0, stream>>>(parts, head_b, out);
}
